// FlexAttentionLayer_60851096649963
// MI455X (gfx1250) — matrix operand order repaired and checked by reading, not compiled, not run on hardware
//
#include <hip/hip_runtime.h>

// Problem constants (match the reference)
#define BB 2
#define TT 2048
#define CC 512
#define HH 8
#define DD 64
#define RR 64

typedef __attribute__((ext_vector_type(16))) __bf16 v16bf;
typedef __attribute__((ext_vector_type(8)))  __bf16 v8bf;
typedef __attribute__((ext_vector_type(8)))  float  v8f;
typedef __attribute__((ext_vector_type(4)))  unsigned int u32x4;
typedef __attribute__((ext_vector_type(8)))  int          i32x8;
typedef __attribute__((ext_vector_type(4)))  int          i32x4;

union ABu { v16bf v; v8bf h[2]; };

__device__ __forceinline__ unsigned short f2bf(float f) {
  unsigned u = __float_as_uint(f);
  u += 0x7FFFu + ((u >> 16) & 1u);        // round-to-nearest-even
  return (unsigned short)(u >> 16);
}
__device__ __forceinline__ v8bf ld8(const unsigned short* p) {
  return *reinterpret_cast<const v8bf*>(p);
}
__device__ __forceinline__ v16bf ld16(const unsigned short* p) {
  return *reinterpret_cast<const v16bf*>(p);
}

// ---------------------------------------------------------------------------
// 1) f32 -> bf16 conversion
// ---------------------------------------------------------------------------
__global__ __launch_bounds__(256) void cvt_bf16_kernel(const float* __restrict__ in,
                                                       unsigned short* __restrict__ out,
                                                       int n) {
  int i = blockIdx.x * 256 + threadIdx.x;
  if (i < n) out[i] = f2bf(in[i]);
}

// ---------------------------------------------------------------------------
// 2) QKV GEMM: qkv[m, n] = sum_k x[m,k] * w_attn[n,k]   (M=4096, N=1536, K=512)
//    One 16x64 output strip per wave (4 WMMA tiles, A operand reused 4x).
//    All B tiles loaded before the WMMA burst so loads overlap WMMA issue.
// ---------------------------------------------------------------------------
__global__ __launch_bounds__(256) void qkv_gemm_kernel(const unsigned short* __restrict__ xb,
                                                       const unsigned short* __restrict__ wab,
                                                       unsigned short* __restrict__ qb,
                                                       unsigned short* __restrict__ kb,
                                                       unsigned short* __restrict__ vtb) {
  const int lane = threadIdx.x & 31;
  const int wv   = threadIdx.x >> 5;
  const int tile = blockIdx.x * 8 + wv;            // 256 * 24 strips total
  const int mt = tile / 24, ng = tile % 24;        // 16 rows x 64 cols per strip
  const int ln = lane & 15, hi = lane >> 4;

  const unsigned short* arow  = xb + (size_t)(mt * 16 + ln) * CC;
  const unsigned short* brow0 = wab + (size_t)(ng * 64 + 0 * 16 + ln) * CC;
  const unsigned short* brow1 = wab + (size_t)(ng * 64 + 1 * 16 + ln) * CC;
  const unsigned short* brow2 = wab + (size_t)(ng * 64 + 2 * 16 + ln) * CC;
  const unsigned short* brow3 = wab + (size_t)(ng * 64 + 3 * 16 + ln) * CC;

  v8f acc[4] = {};
#pragma unroll 2
  for (int k = 0; k < CC; k += 32) {
    ABu a;
    a.h[0] = ld8(arow + k + hi * 16);              // K = hi*16 .. +7
    a.h[1] = ld8(arow + k + hi * 16 + 8);          // K = hi*16+8 .. +15
    v16bf b0 = ld16(brow0 + k + hi * 16);          // lane=col, K = hi*16 .. +15
    v16bf b1 = ld16(brow1 + k + hi * 16);
    v16bf b2 = ld16(brow2 + k + hi * 16);
    v16bf b3 = ld16(brow3 + k + hi * 16);
    acc[0] = __builtin_amdgcn_wmma_f32_16x16x32_bf16(false, a.v, false, b0,
                                                     (short)0, acc[0], false, false);
    acc[1] = __builtin_amdgcn_wmma_f32_16x16x32_bf16(false, a.v, false, b1,
                                                     (short)0, acc[1], false, false);
    acc[2] = __builtin_amdgcn_wmma_f32_16x16x32_bf16(false, a.v, false, b2,
                                                     (short)0, acc[2], false, false);
    acc[3] = __builtin_amdgcn_wmma_f32_16x16x32_bf16(false, a.v, false, b3,
                                                     (short)0, acc[3], false, false);
  }

#pragma unroll
  for (int j = 0; j < 4; ++j) {
    const int gn = ng * 64 + j * 16 + ln;          // output column (uniform over r)
#pragma unroll
    for (int r = 0; r < 8; ++r) {
      const int grow = mt * 16 + r + hi * 8;       // global row in [0, B*T)
      const int b = grow >> 11, t = grow & (TT - 1);
      const float v = acc[j][r];
      if (gn < CC) {                               // Q (fold 1/sqrt(D) = 0.125)
        const int h = gn >> 6, d = gn & 63;
        qb[(((size_t)(b * HH + h)) * TT + t) * DD + d] = f2bf(v * 0.125f);
      } else if (gn < 2 * CC) {                    // K
        const int c2 = gn - CC, h = c2 >> 6, d = c2 & 63;
        kb[(((size_t)(b * HH + h)) * TT + t) * DD + d] = f2bf(v);
      } else {                                     // V, transposed [B,H,D,T]
        const int c2 = gn - 2 * CC, h = c2 >> 6, d = c2 & 63;
        vtb[(((size_t)(b * HH + h)) * DD + d) * TT + t] = f2bf(v);
      }
    }
  }
}

// ---------------------------------------------------------------------------
// 3) Flash attention with relation-bias gather.
//    Block = 256 threads = 8 waves = 8 heads of one (b, 16-row q tile).
//    The shared 16x32 relation tile is DMA'd into LDS by the Tensor Data
//    Mover, double-buffered: wave 0 issues chunk c+1's TDM before computing
//    chunk c, and waits TENSORcnt only at the end-of-iteration barrier.
// ---------------------------------------------------------------------------
__global__ __launch_bounds__(256) void flex_attn_kernel(const unsigned short* __restrict__ qb,
                                                        const unsigned short* __restrict__ kb,
                                                        const unsigned short* __restrict__ vtb,
                                                        const int*   __restrict__ relation,
                                                        const float* __restrict__ rel_pe,
                                                        unsigned short* __restrict__ yb) {
  __shared__ __align__(64) unsigned short plds[8 * 16 * 32]; // per-wave 16x32 bf16 P tile
  __shared__ __align__(16) int relLds[2 * 16 * 32];          // double-buffered relation tile

  const int lane = threadIdx.x & 31;
  const int h    = threadIdx.x >> 5;               // wave id == head
  const int ln   = lane & 15, hi = lane >> 4;
  const int b    = blockIdx.x / (TT / 16);
  const int qrow = (blockIdx.x % (TT / 16)) * 16;
  const int NC   = TT / 32;                        // kv chunks

  const unsigned short* qp  = qb  + ((size_t)(b * HH + h)) * TT * DD;
  const unsigned short* kp  = kb  + ((size_t)(b * HH + h)) * TT * DD;
  const unsigned short* vtp = vtb + ((size_t)(b * HH + h)) * DD * TT;
  const int*   relb = relation + (size_t)b * TT * TT;
  const float* peb  = rel_pe   + (size_t)b * TT * RR * HH;
  unsigned short* lbase = plds + h * 512;

  // TDM descriptor pieces (wave 0 only uses them).
  const unsigned ldsOff = (unsigned)(uintptr_t)(void*)relLds;  // low 32 bits = LDS offset
  // D# group1: data_size=4B(code 2), tile = 32 x 16, tensor_dim0/1 = 2048,
  // tensor_dim0_stride = 2048, no padding / multicast / iterate.
  const i32x8 g1 = { 0x00020000, 0x08000000, 0x08000000, 0x00200000,
                     16, 2048, 0, 0 };
  const i32x4 gz4 = { 0, 0, 0, 0 };
  const i32x8 gz8 = { 0, 0, 0, 0, 0, 0, 0, 0 };

  auto tdm_issue = [&](int cc, int buf) {
    const unsigned long long ga =
        (unsigned long long)(uintptr_t)(relb + (size_t)qrow * TT + cc * 32);
    u32x4 g0;
    g0[0] = 1u;                                    // count=1, user descriptor
    g0[1] = ldsOff + (unsigned)(buf * 2048);       // lds_addr (bytes)
    g0[2] = (unsigned)ga;                          // global_addr[31:0]
    g0[3] = ((unsigned)(ga >> 32) & 0x01FFFFFFu)   // global_addr[56:32]
            | 0x80000000u;                         // type = 2 ("image")
    __builtin_amdgcn_tensor_load_to_lds(g0, g1, gz4, gz4, gz8, 0);
  };

  // Q A-operands (held for the whole kv loop)
  const unsigned short* qrp = qp + (size_t)(qrow + ln) * DD;
  ABu aq0, aq1;
  aq0.h[0] = ld8(qrp + hi * 16);       aq0.h[1] = ld8(qrp + 8 + hi * 16);
  aq1.h[0] = ld8(qrp + 32 + hi * 16);  aq1.h[1] = ld8(qrp + 40 + hi * 16);

  const float NEG_INF = -__builtin_inff();
  float Mrow[8], Lrow[8];
  v8f o0 = {}, o1 = {}, o2 = {}, o3 = {};
#pragma unroll
  for (int r = 0; r < 8; ++r) { Mrow[r] = -1e30f; Lrow[r] = 0.0f; }

  // Prime buffer 0 with chunk 0's relation tile.
  if (h == 0) {
    tdm_issue(0, 0);
    __builtin_amdgcn_s_wait_tensorcnt(0);
  }
  __syncthreads();

#pragma unroll 1
  for (int c = 0; c < NC; ++c) {
    const int kvb = c * 32;
    const int* relBuf = relLds + (c & 1) * 512;

    // Prefetch next chunk's relation tile into the other buffer (overlapped).
    if (h == 0 && c + 1 < NC) tdm_issue(c + 1, (c + 1) & 1);

    // ---- S = Q K^T (two 16x16 tiles); K tiles batched before WMMA burst ----
    const unsigned short* kr0 = kp + (size_t)(kvb + ln) * DD;
    const unsigned short* kr1 = kp + (size_t)(kvb + 16 + ln) * DD;
    v16bf bk00 = ld16(kr0 + hi * 16);              // tile0, d = 0..31
    v16bf bk01 = ld16(kr0 + 32 + hi * 16);         // tile0, d = 32..63
    v16bf bk10 = ld16(kr1 + hi * 16);              // tile1, d = 0..31
    v16bf bk11 = ld16(kr1 + 32 + hi * 16);         // tile1, d = 32..63
    v8f s[2];
    s[0] = (v8f){};
    s[0] = __builtin_amdgcn_wmma_f32_16x16x32_bf16(false, aq0.v, false, bk00,
                                                   (short)0, s[0], false, false);
    s[0] = __builtin_amdgcn_wmma_f32_16x16x32_bf16(false, aq1.v, false, bk01,
                                                   (short)0, s[0], false, false);
    s[1] = (v8f){};
    s[1] = __builtin_amdgcn_wmma_f32_16x16x32_bf16(false, aq0.v, false, bk10,
                                                   (short)0, s[1], false, false);
    s[1] = __builtin_amdgcn_wmma_f32_16x16x32_bf16(false, aq1.v, false, bk11,
                                                   (short)0, s[1], false, false);

    // ---- bias + mask from the LDS relation tile ----
#pragma unroll
    for (int j = 0; j < 2; ++j) {
#pragma unroll
      for (int r = 0; r < 8; ++r) {
        const int m = r + hi * 8;
        const int q = qrow + m;
        const int rel = relBuf[m * 32 + j * 16 + ln];      // ds_load, shared tile
        const int rc  = rel < 0 ? 0 : rel;
        const float bias = peb[((size_t)q * RR + rc) * HH + h];
        s[j][r] = (rel < 0) ? NEG_INF : (s[j][r] + bias);
      }
    }

    // ---- online softmax over the 32 kv columns of this chunk ----
#pragma unroll
    for (int r = 0; r < 8; ++r) {
      float v0 = s[0][r], v1 = s[1][r];
      float mx = fmaxf(v0, v1);
#pragma unroll
      for (int off = 8; off > 0; off >>= 1) mx = fmaxf(mx, __shfl_xor(mx, off, 16));
      const float mnew  = fmaxf(Mrow[r], mx);      // clamped by -1e30 init
      const float alpha = __expf(Mrow[r] - mnew);
      Mrow[r] = mnew;
      v0 = __expf(v0 - mnew);
      v1 = __expf(v1 - mnew);
      float rs = v0 + v1;
#pragma unroll
      for (int off = 8; off > 0; off >>= 1) rs += __shfl_xor(rs, off, 16);
      Lrow[r] = Lrow[r] * alpha + rs;
      o0[r] *= alpha; o1[r] *= alpha; o2[r] *= alpha; o3[r] *= alpha;
      const int m = r + hi * 8;                    // P tile row-major in LDS
      lbase[m * 32 + ln]      = f2bf(v0);
      lbase[m * 32 + 16 + ln] = f2bf(v1);
    }
    asm volatile("s_wait_dscnt 0" ::: "memory");

    // ---- reload P in A-operand layout, then O += P V (V tiles batched) ----
    const unsigned short* prp = lbase + ln * 32;
    ABu ap;
    ap.h[0] = ld8(prp + hi * 16);
    ap.h[1] = ld8(prp + 8 + hi * 16);

    v16bf bv0 = ld16(vtp + (size_t)(0 * 16 + ln) * TT + kvb + hi * 16);
    v16bf bv1 = ld16(vtp + (size_t)(1 * 16 + ln) * TT + kvb + hi * 16);
    v16bf bv2 = ld16(vtp + (size_t)(2 * 16 + ln) * TT + kvb + hi * 16);
    v16bf bv3 = ld16(vtp + (size_t)(3 * 16 + ln) * TT + kvb + hi * 16);
    o0 = __builtin_amdgcn_wmma_f32_16x16x32_bf16(false, ap.v, false, bv0,
                                                 (short)0, o0, false, false);
    o1 = __builtin_amdgcn_wmma_f32_16x16x32_bf16(false, ap.v, false, bv1,
                                                 (short)0, o1, false, false);
    o2 = __builtin_amdgcn_wmma_f32_16x16x32_bf16(false, ap.v, false, bv2,
                                                 (short)0, o2, false, false);
    o3 = __builtin_amdgcn_wmma_f32_16x16x32_bf16(false, ap.v, false, bv3,
                                                 (short)0, o3, false, false);

    // Next chunk's relation tile must have landed; everyone done with relBuf.
    if (h == 0 && c + 1 < NC) __builtin_amdgcn_s_wait_tensorcnt(0);
    __syncthreads();
  }

  // ---- epilogue: y[b, t, h*64 + d] = O / L ----
#pragma unroll
  for (int r = 0; r < 8; ++r) {
    const int m = r + hi * 8;
    const float inv = 1.0f / fmaxf(Lrow[r], 1e-20f);
    unsigned short* yr = yb + ((size_t)(b * TT + qrow + m)) * CC + h * DD;
    yr[0 * 16 + ln] = f2bf(o0[r] * inv);
    yr[1 * 16 + ln] = f2bf(o1[r] * inv);
    yr[2 * 16 + ln] = f2bf(o2[r] * inv);
    yr[3 * 16 + ln] = f2bf(o3[r] * inv);
  }
}

// ---------------------------------------------------------------------------
// 4) Output projection: out[m, n] = sum_k y[m,k] * w_proj[n,k]  (f32 out)
//    16x64 strip per wave, A reused across 4 WMMA tiles, batched B loads.
// ---------------------------------------------------------------------------
__global__ __launch_bounds__(256) void proj_gemm_kernel(const unsigned short* __restrict__ ybm,
                                                        const unsigned short* __restrict__ wpb,
                                                        float* __restrict__ out) {
  const int lane = threadIdx.x & 31;
  const int wv   = threadIdx.x >> 5;
  const int tile = blockIdx.x * 8 + wv;            // 256 * 8 strips
  const int mt = tile / 8, ng = tile % 8;
  const int ln = lane & 15, hi = lane >> 4;

  const unsigned short* arow  = ybm + (size_t)(mt * 16 + ln) * CC;
  const unsigned short* brow0 = wpb + (size_t)(ng * 64 + 0 * 16 + ln) * CC;
  const unsigned short* brow1 = wpb + (size_t)(ng * 64 + 1 * 16 + ln) * CC;
  const unsigned short* brow2 = wpb + (size_t)(ng * 64 + 2 * 16 + ln) * CC;
  const unsigned short* brow3 = wpb + (size_t)(ng * 64 + 3 * 16 + ln) * CC;

  v8f acc[4] = {};
#pragma unroll 2
  for (int k = 0; k < CC; k += 32) {
    ABu a;
    a.h[0] = ld8(arow + k + hi * 16);
    a.h[1] = ld8(arow + k + hi * 16 + 8);
    v16bf b0 = ld16(brow0 + k + hi * 16);
    v16bf b1 = ld16(brow1 + k + hi * 16);
    v16bf b2 = ld16(brow2 + k + hi * 16);
    v16bf b3 = ld16(brow3 + k + hi * 16);
    acc[0] = __builtin_amdgcn_wmma_f32_16x16x32_bf16(false, a.v, false, b0,
                                                     (short)0, acc[0], false, false);
    acc[1] = __builtin_amdgcn_wmma_f32_16x16x32_bf16(false, a.v, false, b1,
                                                     (short)0, acc[1], false, false);
    acc[2] = __builtin_amdgcn_wmma_f32_16x16x32_bf16(false, a.v, false, b2,
                                                     (short)0, acc[2], false, false);
    acc[3] = __builtin_amdgcn_wmma_f32_16x16x32_bf16(false, a.v, false, b3,
                                                     (short)0, acc[3], false, false);
  }

#pragma unroll
  for (int j = 0; j < 4; ++j) {
    const int gn = ng * 64 + j * 16 + ln;
#pragma unroll
    for (int r = 0; r < 8; ++r) {
      const int grow = mt * 16 + r + hi * 8;
      out[(size_t)grow * CC + gn] = acc[j][r];
    }
  }
}

// ---------------------------------------------------------------------------
// Host launcher
// ---------------------------------------------------------------------------
extern "C" void kernel_launch(void* const* d_in, const int* in_sizes, int n_in,
                              void* d_out, int out_size, void* d_ws, size_t ws_size,
                              hipStream_t stream) {
  const float* x      = (const float*)d_in[0];   // [B,T,C]
  const float* w_attn = (const float*)d_in[1];   // [3C,C]
  const float* w_proj = (const float*)d_in[2];   // [C,C]
  const float* rel_pe = (const float*)d_in[3];   // [B,T,R,H]
  const int*   relation = (const int*)d_in[4];   // [B,T,T]
  float* out = (float*)d_out;

  char* ws = (char*)d_ws;
  size_t o = 0;
  auto carve = [&](size_t bytes) { char* p = ws + o; o = (o + bytes + 255) & ~(size_t)255; return p; };
  const size_t nX  = (size_t)BB * TT * CC;        // 2,097,152
  const size_t nWA = (size_t)3 * CC * CC;         //   786,432
  const size_t nWP = (size_t)CC * CC;             //   262,144
  unsigned short* xb  = (unsigned short*)carve(nX * 2);
  unsigned short* wab = (unsigned short*)carve(nWA * 2);
  unsigned short* wpb = (unsigned short*)carve(nWP * 2);
  unsigned short* qb  = (unsigned short*)carve(nX * 2);   // [B,H,T,D]
  unsigned short* kb  = (unsigned short*)carve(nX * 2);   // [B,H,T,D]
  unsigned short* vtb = (unsigned short*)carve(nX * 2);   // [B,H,D,T]
  unsigned short* yb  = (unsigned short*)carve(nX * 2);   // [B,T,C]

  cvt_bf16_kernel<<<(int)((nX  + 255) / 256), 256, 0, stream>>>(x,      xb,  (int)nX);
  cvt_bf16_kernel<<<(int)((nWA + 255) / 256), 256, 0, stream>>>(w_attn, wab, (int)nWA);
  cvt_bf16_kernel<<<(int)((nWP + 255) / 256), 256, 0, stream>>>(w_proj, wpb, (int)nWP);

  qkv_gemm_kernel<<<(256 * 24) / 8, 256, 0, stream>>>(xb, wab, qb, kb, vtb);
  flex_attn_kernel<<<BB * (TT / 16), 256, 0, stream>>>(qb, kb, vtb, relation, rel_pe, yb);
  proj_gemm_kernel<<<(256 * 8) / 8, 256, 0, stream>>>(yb, wpb, out);
}
